// PointSetAttention_29832842838756
// MI455X (gfx1250) — matrix operand order repaired and checked by reading, not compiled, not run on hardware
//
#include <hip/hip_runtime.h>

#define NN 20000
#define EE 320000
#define CC 256

typedef __attribute__((ext_vector_type(16))) __bf16 v16bf;
typedef __attribute__((ext_vector_type(8)))  float  v8f;
typedef __attribute__((ext_vector_type(4)))  unsigned int u32x4;

union Frag { v16bf v; u32x4 u[2]; };

__device__ __forceinline__ unsigned short f2bf(float f) {
    unsigned u = __float_as_uint(f);
    u += 0x7FFFu + ((u >> 16) & 1u);           // round-to-nearest-even
    return (unsigned short)(u >> 16);
}
__device__ __forceinline__ float bflo(unsigned u) { return __uint_as_float(u << 16); }
__device__ __forceinline__ float bfhi(unsigned u) { return __uint_as_float(u & 0xFFFF0000u); }

__device__ __forceinline__ void atomicMaxF(float* a, float v) {
    // bit-trick float max; m initialized to -inf (works for mixed signs)
    if (v >= 0.f) atomicMax((int*)a, __float_as_int(v));
    else          atomicMin((unsigned int*)a, __float_as_uint(v));
}

// ---------------- elementwise helpers ----------------

__global__ __launch_bounds__(256) void fill_f32(float* p, float v, int n) {
    int t = blockIdx.x * 256 + threadIdx.x;
    if (t < n) p[t] = v;
}

__global__ __launch_bounds__(256) void f32_to_bf16(const float* __restrict__ s,
                                                   unsigned short* __restrict__ d, int n) {
    int t = blockIdx.x * 256 + threadIdx.x;
    if (t < n) d[t] = f2bf(s[t]);
}

// W [K,N] fp32 row-major -> WT [N,K] bf16
__global__ __launch_bounds__(256) void transpose_to_bf16(const float* __restrict__ W,
                                                         unsigned short* __restrict__ WT,
                                                         int K, int N) {
    int t = blockIdx.x * 256 + threadIdx.x;
    if (t >= K * N) return;
    int k = t / N, n = t - k * N;
    WT[(size_t)n * K + k] = f2bf(W[t]);
}

// Pre-reduce Wrpe [3,256] over each head's 64 channels -> rpeH[12] + bias[4]
__global__ void rpe_reduce(const float* __restrict__ Wrpe, const float* __restrict__ brpe,
                           float* __restrict__ rpeH) {
    int t = threadIdx.x;
    if (t < 12) {
        int p = t >> 2, h = t & 3;
        float s = 0.f;
        for (int d = 0; d < 64; ++d) s += Wrpe[p * 256 + h * 64 + d];
        rpeH[t] = s;
    } else if (t < 16) {
        int h = t - 12;
        float s = 0.f;
        for (int d = 0; d < 64; ++d) s += brpe[h * 64 + d];
        rpeH[t] = s;
    }
}

// ---------------- WMMA GEMM: C[M,N] = A[M,K](bf16) @ WT[N,K]^T(bf16) + bias ----------------
// One wave computes a 16x64 output strip (4 accumulators). A-fragment reused 4x.

template<bool BF16OUT>
__global__ __launch_bounds__(256) void wmma_gemm_kernel(
    const unsigned short* __restrict__ A, const unsigned short* __restrict__ WT,
    const float* __restrict__ bias, void* __restrict__ out, int M, int N, int K) {

    const int wave = (int)((blockIdx.x * 256u + threadIdx.x) >> 5);
    const int lane = threadIdx.x & 31;
    const int rowTiles = M >> 4;
    const int colGrps  = N >> 6;
    if (wave >= rowTiles * colGrps) return;
    const int rowT = wave % rowTiles;
    const int colG = wave / rowTiles;
    const int half = lane >> 4;
    const int lsub = lane & 15;

    const unsigned short* Ar = A + (size_t)(rowT * 16 + lsub) * K;
    const int colBase = colG * 64;
    const unsigned short* W0 = WT + (size_t)(colBase +  0 + lsub) * K;
    const unsigned short* W1 = WT + (size_t)(colBase + 16 + lsub) * K;
    const unsigned short* W2 = WT + (size_t)(colBase + 32 + lsub) * K;
    const unsigned short* W3 = WT + (size_t)(colBase + 48 + lsub) * K;

    v8f c0 = {}, c1 = {}, c2 = {}, c3 = {};
    for (int kb = 0; kb < K; kb += 32) {
        Frag a, b0, b1, b2, b3;
        // A 16x32 bf16 layout: lanes 0-15 -> K0-15 (v0-7); lanes 16-31 -> K16-31 (contiguous, matches B)
        a.u[0] = *(const u32x4*)(Ar + kb + half * 16);
        a.u[1] = *(const u32x4*)(Ar + kb + half * 16 + 8);
        // B 32x16 bf16: lanes 0-15 -> K0-15 in v0-7, lanes 16-31 -> K16-31 (contiguous in WT rows)
        const int bo = kb + half * 16;
        b0.u[0] = *(const u32x4*)(W0 + bo); b0.u[1] = *(const u32x4*)(W0 + bo + 8);
        b1.u[0] = *(const u32x4*)(W1 + bo); b1.u[1] = *(const u32x4*)(W1 + bo + 8);
        b2.u[0] = *(const u32x4*)(W2 + bo); b2.u[1] = *(const u32x4*)(W2 + bo + 8);
        b3.u[0] = *(const u32x4*)(W3 + bo); b3.u[1] = *(const u32x4*)(W3 + bo + 8);
        c0 = __builtin_amdgcn_wmma_f32_16x16x32_bf16(false, a.v, false, b0.v, (short)0, c0, false, false);
        c1 = __builtin_amdgcn_wmma_f32_16x16x32_bf16(false, a.v, false, b1.v, (short)0, c1, false, false);
        c2 = __builtin_amdgcn_wmma_f32_16x16x32_bf16(false, a.v, false, b2.v, (short)0, c2, false, false);
        c3 = __builtin_amdgcn_wmma_f32_16x16x32_bf16(false, a.v, false, b3.v, (short)0, c3, false, false);
    }

    // C layout: VGPR r -> row rowT*16 + r + half*8 ; col = colBase + tile*16 + lsub
    const int rbase = rowT * 16 + half * 8;
    #pragma unroll
    for (int t = 0; t < 4; ++t) {
        const v8f& cc = (t == 0) ? c0 : (t == 1) ? c1 : (t == 2) ? c2 : c3;
        const int col = colBase + t * 16 + lsub;
        const float bv = bias[col];
        #pragma unroll
        for (int r = 0; r < 8; ++r) {
            const float val = cc[r] + bv;
            const size_t idx = (size_t)(rbase + r) * (size_t)N + col;
            if (BF16OUT) ((unsigned short*)out)[idx] = f2bf(val);
            else         ((float*)out)[idx] = val;
        }
    }
}

// ---------------- edge attention kernels (wave32 per edge) ----------------

// s[e,h] = sum_d q[g0[e],h,d]*k[g1[e],h,d] (+ rpe head score); atomic segment-max into m
__global__ __launch_bounds__(256) void edge_score(
    const unsigned short* __restrict__ q, const unsigned short* __restrict__ k, int ld,
    const int* __restrict__ g, int E,
    const float* __restrict__ coord, const float* __restrict__ rpeH,
    float* __restrict__ sbuf, float* __restrict__ mbuf) {

    int w = (int)((blockIdx.x * 256u + threadIdx.x) >> 5);
    if (w >= E) return;
    int lane = threadIdx.x & 31;
    int dst = g[w];
    int src = g[E + w];
    const u32x4 qv = *(const u32x4*)(q + (size_t)dst * ld + lane * 8);
    const u32x4 kv = *(const u32x4*)(k + (size_t)src * ld + lane * 8);
    float p = 0.f;
    p += bflo(qv[0]) * bflo(kv[0]) + bfhi(qv[0]) * bfhi(kv[0]);
    p += bflo(qv[1]) * bflo(kv[1]) + bfhi(qv[1]) * bfhi(kv[1]);
    p += bflo(qv[2]) * bflo(kv[2]) + bfhi(qv[2]) * bfhi(kv[2]);
    p += bflo(qv[3]) * bflo(kv[3]) + bfhi(qv[3]) * bfhi(kv[3]);
    // reduce within 8-lane head groups (lanes 8h..8h+7 cover d 0..63 of head h)
    p += __shfl_xor(p, 1);
    p += __shfl_xor(p, 2);
    p += __shfl_xor(p, 4);
    if ((lane & 7) == 0) {
        int h = lane >> 3;
        if (coord) {
            float d0 = coord[3 * dst + 0] - coord[3 * src + 0];
            float d1 = coord[3 * dst + 1] - coord[3 * src + 1];
            float d2 = coord[3 * dst + 2] - coord[3 * src + 2];
            p += d0 * rpeH[h] + d1 * rpeH[4 + h] + d2 * rpeH[8 + h] + rpeH[12 + h];
        }
        sbuf[(size_t)w * 4 + h] = p;
        atomicMaxF(mbuf + (size_t)dst * 4 + h, p);
    }
}

// e = exp(s - m[dst]);  z[dst] += e  (thread per edge*head)
__global__ __launch_bounds__(256) void edge_expsum(
    float* __restrict__ sbuf, const float* __restrict__ mbuf, float* __restrict__ zbuf,
    const int* __restrict__ g, int E) {
    int t = blockIdx.x * 256 + threadIdx.x;
    if (t >= E * 4) return;
    int e = t >> 2, h = t & 3;
    int dst = g[e];
    float ex = __expf(sbuf[t] - mbuf[(size_t)dst * 4 + h]);
    sbuf[t] = ex;
    atomicAdd(zbuf + (size_t)dst * 4 + h, ex);
}

// out[dst, colOff + h*64 + d] += (e/z[dst]) * v[src, h*64 + d]   (wave per edge)
__global__ __launch_bounds__(256) void edge_scatter(
    const float* __restrict__ sbuf, const float* __restrict__ zbuf,
    const unsigned short* __restrict__ v, int ld,
    const int* __restrict__ g, int E,
    float* __restrict__ out, int outStride, int colOff) {

    int w = (int)((blockIdx.x * 256u + threadIdx.x) >> 5);
    if (w >= E) return;
    int lane = threadIdx.x & 31;
    int h = lane >> 3;
    int doff = (lane & 7) * 8;
    int dst = g[w];
    int src = g[E + w];
    float a = sbuf[(size_t)w * 4 + h] / zbuf[(size_t)dst * 4 + h];
    const u32x4 vv = *(const u32x4*)(v + (size_t)src * ld + h * 64 + doff);
    float* o = out + (size_t)dst * outStride + colOff + h * 64 + doff;
    atomicAdd(o + 0, a * bflo(vv[0]));
    atomicAdd(o + 1, a * bfhi(vv[0]));
    atomicAdd(o + 2, a * bflo(vv[1]));
    atomicAdd(o + 3, a * bfhi(vv[1]));
    atomicAdd(o + 4, a * bflo(vv[2]));
    atomicAdd(o + 5, a * bfhi(vv[2]));
    atomicAdd(o + 6, a * bflo(vv[3]));
    atomicAdd(o + 7, a * bfhi(vv[3]));
}

// ---------------- host side ----------------

extern "C" void kernel_launch(void* const* d_in, const int* in_sizes, int n_in,
                              void* d_out, int out_size, void* d_ws, size_t ws_size,
                              hipStream_t stream) {
    const float* feat1  = (const float*)d_in[0];
    const float* coord1 = (const float*)d_in[1];
    const float* feat2  = (const float*)d_in[2];
    const float* coord2 = (const float*)d_in[3];
    const float* Wqkv1  = (const float*)d_in[4];
    const float* bqkv1  = (const float*)d_in[5];
    const float* Wqkv2  = (const float*)d_in[6];
    const float* bqkv2  = (const float*)d_in[7];
    const float* Wproj1 = (const float*)d_in[8];
    const float* bproj1 = (const float*)d_in[9];
    const float* Wproj2 = (const float*)d_in[10];
    const float* bproj2 = (const float*)d_in[11];
    const float* Wrpe   = (const float*)d_in[12];
    const float* brpe   = (const float*)d_in[13];
    const int* graph1   = (const int*)d_in[14];
    const int* graph2   = (const int*)d_in[15];
    const int* graph12  = (const int*)d_in[16];
    const int* graph21  = (const int*)d_in[17];

    char* ws = (char*)d_ws;
    size_t off = 0;
    auto alloc = [&](size_t bytes) { char* p = ws + off; off += (bytes + 255) & ~(size_t)255; return p; };

    unsigned short* fbf1   = (unsigned short*)alloc((size_t)NN * CC * 2);
    unsigned short* fbf2   = (unsigned short*)alloc((size_t)NN * CC * 2);
    unsigned short* WT1    = (unsigned short*)alloc((size_t)1536 * 256 * 2);
    unsigned short* WT2    = (unsigned short*)alloc((size_t)1536 * 256 * 2);
    unsigned short* PT1    = (unsigned short*)alloc((size_t)256 * 512 * 2);
    unsigned short* PT2    = (unsigned short*)alloc((size_t)256 * 512 * 2);
    float*          rpeH   = (float*)alloc(16 * 4);
    unsigned short* qkvbf1 = (unsigned short*)alloc((size_t)NN * 1536 * 2);
    unsigned short* qkvbf2 = (unsigned short*)alloc((size_t)NN * 1536 * 2);
    float*          cat1   = (float*)alloc((size_t)NN * 512 * 4);
    float*          cat2   = (float*)alloc((size_t)NN * 512 * 4);
    unsigned short* catbf1 = (unsigned short*)alloc((size_t)NN * 512 * 2);
    unsigned short* catbf2 = (unsigned short*)alloc((size_t)NN * 512 * 2);
    float*          sbuf   = (float*)alloc((size_t)EE * 4 * 4);
    float*          mbuf   = (float*)alloc((size_t)NN * 4 * 4);
    float*          zbuf   = (float*)alloc((size_t)NN * 4 * 4);

    auto nb = [](int n) { return (n + 255) / 256; };

    // 1) convert activations, transpose weights, reduce rpe
    f32_to_bf16<<<nb(NN * CC), 256, 0, stream>>>(feat1, fbf1, NN * CC);
    f32_to_bf16<<<nb(NN * CC), 256, 0, stream>>>(feat2, fbf2, NN * CC);
    transpose_to_bf16<<<nb(256 * 1536), 256, 0, stream>>>(Wqkv1, WT1, 256, 1536);
    transpose_to_bf16<<<nb(256 * 1536), 256, 0, stream>>>(Wqkv2, WT2, 256, 1536);
    transpose_to_bf16<<<nb(512 * 256), 256, 0, stream>>>(Wproj1, PT1, 512, 256);
    transpose_to_bf16<<<nb(512 * 256), 256, 0, stream>>>(Wproj2, PT2, 512, 256);
    rpe_reduce<<<1, 64, 0, stream>>>(Wrpe, brpe, rpeH);

    // 2) QKV GEMMs  (M=20000, N=1536, K=256) -> bf16
    {
        int waves = (NN / 16) * (1536 / 64);
        wmma_gemm_kernel<true><<<(waves + 7) / 8, 256, 0, stream>>>(fbf1, WT1, bqkv1, qkvbf1, NN, 1536, 256);
        wmma_gemm_kernel<true><<<(waves + 7) / 8, 256, 0, stream>>>(fbf2, WT2, bqkv2, qkvbf2, NN, 1536, 256);
    }

    // 3) zero attention accumulators
    fill_f32<<<nb(NN * 512), 256, 0, stream>>>(cat1, 0.f, NN * 512);
    fill_f32<<<nb(NN * 512), 256, 0, stream>>>(cat2, 0.f, NN * 512);

    // 4) four scatter-softmax attention passes
    struct Pass {
        const unsigned short *q, *k, *v;
        const int* g;
        const float* coord;
        float* out;
        int colOff;
    } passes[4] = {
        { qkvbf1 + 0,   qkvbf1 + 256,  qkvbf1 + 512,  graph1,  coord1, cat1, 0   },  // feat11
        { qkvbf2 + 768, qkvbf2 + 1024, qkvbf2 + 1280, graph2,  coord2, cat2, 0   },  // feat22
        { qkvbf1 + 768, qkvbf2 + 256,  qkvbf2 + 512,  graph21, nullptr, cat1, 256 }, // feat12
        { qkvbf2 + 0,   qkvbf1 + 1024, qkvbf1 + 1280, graph12, nullptr, cat2, 256 }, // feat21
    };
    const int edgeBlocks = EE / 8;  // wave32 per edge, 8 waves/block
    for (int p = 0; p < 4; ++p) {
        fill_f32<<<nb(NN * 4), 256, 0, stream>>>(mbuf, -INFINITY, NN * 4);
        fill_f32<<<nb(NN * 4), 256, 0, stream>>>(zbuf, 0.f, NN * 4);
        edge_score<<<edgeBlocks, 256, 0, stream>>>(passes[p].q, passes[p].k, 1536,
                                                   passes[p].g, EE, passes[p].coord, rpeH,
                                                   sbuf, mbuf);
        edge_expsum<<<nb(EE * 4), 256, 0, stream>>>(sbuf, mbuf, zbuf, passes[p].g, EE);
        edge_scatter<<<edgeBlocks, 256, 0, stream>>>(sbuf, zbuf, passes[p].v, 1536,
                                                     passes[p].g, EE, passes[p].out, 512,
                                                     passes[p].colOff);
    }

    // 5) output projections  (M=20000, N=256, K=512) -> f32 directly into d_out
    f32_to_bf16<<<nb(NN * 512), 256, 0, stream>>>(cat1, catbf1, NN * 512);
    f32_to_bf16<<<nb(NN * 512), 256, 0, stream>>>(cat2, catbf2, NN * 512);
    {
        int waves = (NN / 16) * (256 / 64);
        float* out1 = (float*)d_out;
        float* out2 = out1 + (size_t)NN * 256;
        wmma_gemm_kernel<false><<<(waves + 7) / 8, 256, 0, stream>>>(catbf1, PT1, bproj1, out1, NN, 256, 512);
        wmma_gemm_kernel<false><<<(waves + 7) / 8, 256, 0, stream>>>(catbf2, PT2, bproj2, out2, NN, 256, 512);
    }
}
